// Megnet_NodeModel_62689342653098
// MI455X (gfx1250) — matrix operand order repaired and checked by reading, not compiled, not run on hardware
//
#include <hip/hip_runtime.h>
#include <hip/hip_bf16.h>

// ---------- types for WMMA ----------
typedef __attribute__((ext_vector_type(16))) __bf16        v16bf;
typedef __attribute__((ext_vector_type(8)))  float         v8f;
typedef __attribute__((ext_vector_type(8)))  unsigned short v8u16;

union Frag16 {            // 16 bf16 = 32 bytes, loaded as two 16B chunks
    v16bf bf;
    v8u16 u[2];
};

__device__ __forceinline__ unsigned short f2bf(float f) {
    unsigned int x = __float_as_uint(f);
    unsigned int r = x + 0x7FFFu + ((x >> 16) & 1u);   // round-to-nearest-even
    return (unsigned short)(r >> 16);
}

// ---------- zero ----------
__global__ void zero_kernel(float* __restrict__ p, long n) {
    long t = (long)blockIdx.x * blockDim.x + threadIdx.x;
    if (t < n) p[t] = 0.0f;
}

// ---------- scatter-mean accumulation ----------
__global__ void scatter_kernel(const int* __restrict__ src,
                               const float* __restrict__ edge_attr,
                               float* __restrict__ sums, float* __restrict__ cnt,
                               int E_) {
    int t = blockIdx.x * blockDim.x + threadIdx.x;
    int e = t >> 2;
    if (e >= E_) return;
    int cg = (t & 3) * 32;
    int s  = src[e];
    const float* ea  = edge_attr + (long)e * 128 + cg;
    float*       dst = sums      + (long)s * 128 + cg;
    #pragma unroll
    for (int j = 0; j < 32; j += 4) {
        float4 v = *(const float4*)(ea + j);
        unsafeAtomicAdd(dst + j + 0, v.x);
        unsafeAtomicAdd(dst + j + 1, v.y);
        unsafeAtomicAdd(dst + j + 2, v.z);
        unsafeAtomicAdd(dst + j + 3, v.w);
    }
    if (cg == 0) unsafeAtomicAdd(cnt + s, 1.0f);
}

// ---------- build comb = [x, v_e, u[batch]] in bf16 ----------
__global__ void comb_kernel(const float* __restrict__ x, const float* __restrict__ sums,
                            const float* __restrict__ cnt, const float* __restrict__ u,
                            const int* __restrict__ batch,
                            unsigned short* __restrict__ comb, int n) {
    long t = (long)blockIdx.x * blockDim.x + threadIdx.x;
    if (t >= (long)n * 384) return;
    int i = (int)(t / 384);
    int j = (int)(t % 384);
    float val;
    if (j < 128) {
        val = x[(long)i * 128 + j];
    } else if (j < 256) {
        float c = cnt[i];
        c = c < 1.0f ? 1.0f : c;
        val = sums[(long)i * 128 + (j - 128)] / c;
    } else {
        val = u[(long)batch[i] * 128 + (j - 256)];
    }
    comb[t] = f2bf(val);
}

// ---------- swizzle weights into WMMA B-fragment layout ----------
// B (K x 16 tile) lane layout: lanes 0-15 hold K 0-7 then 16-23, col = lane;
// lanes 16-31 hold K 8-15 then 24-31, col = lane-16. 16 contiguous bf16/lane.
__global__ void wswz_kernel(const float* __restrict__ W, unsigned short* __restrict__ out,
                            int K) {
    int t = blockIdx.x * blockDim.x + threadIdx.x;
    int total = (K >> 5) * 8 * 512;
    if (t >= total) return;
    int e    =  t        & 15;
    int lane = (t >> 4)  & 31;
    int c    = (t >> 9)  & 7;
    int kc   =  t >> 12;
    int klocal = ((lane < 16) ? 0 : 8) + (e & 7) + ((e & 8) ? 16 : 0);
    int col    = c * 16 + (lane & 15);
    int k      = kc * 32 + klocal;
    out[t] = f2bf(W[(long)k * 128 + col]);
}

// ---------- fused GEMM + bias + ReLU + BN-stat accumulation ----------
__global__ __launch_bounds__(256)
void gemm_bn_kernel(const unsigned short* __restrict__ A, int K,
                    const unsigned short* __restrict__ Wswz,
                    const float* __restrict__ bias,
                    float* __restrict__ H,
                    float* __restrict__ colsum, float* __restrict__ colsumsq,
                    int nrows) {
    __shared__ float lsum[128];
    __shared__ float lsumsq[128];
    int tid = threadIdx.x;
    if (tid < 128) { lsum[tid] = 0.0f; lsumsq[tid] = 0.0f; }
    __syncthreads();

    int wave = tid >> 5;
    int lane = tid & 31;
    int r0   = blockIdx.x * 128 + wave * 16;

    // A fragment addressing (16-bit A 16x32 layout)
    int  arow   = r0 + (lane & 15);
    long arow_c = (arow < nrows) ? (long)arow : (long)(nrows - 1);
    const unsigned short* aptr = A + arow_c * K;
    int  koff0  = (lane < 16) ? 0 : 8;

    v8f acc[8] = {};
    int nkc = K >> 5;
    for (int kc = 0; kc < nkc; ++kc) {
        Frag16 afrag;
        int kb = kc * 32 + koff0;
        afrag.u[0] = *(const v8u16*)(aptr + kb);
        afrag.u[1] = *(const v8u16*)(aptr + kb + 16);
        const unsigned short* wp = Wswz + ((long)(kc * 8) * 32 + lane) * 16;
        #pragma unroll
        for (int c = 0; c < 8; ++c) {
            Frag16 bfrag;
            bfrag.u[0] = *(const v8u16*)(wp);
            bfrag.u[1] = *(const v8u16*)(wp + 8);
            wp += 32 * 16;
            acc[c] = __builtin_amdgcn_wmma_f32_16x16x32_bf16(
                false, afrag.bf, false, bfrag.bf, (short)0, acc[c], false, false);
        }
    }

    // epilogue: bias + ReLU + store + column stats
    int coln  = lane & 15;
    int rbase = r0 + ((lane < 16) ? 0 : 8);
    #pragma unroll
    for (int c = 0; c < 8; ++c) {
        int   col = c * 16 + coln;
        float b   = bias[col];
        float ps = 0.0f, psq = 0.0f;
        #pragma unroll
        for (int v = 0; v < 8; ++v) {
            float h = acc[c][v] + b;
            h = h > 0.0f ? h : 0.0f;
            int row = rbase + v;
            if (row < nrows) {
                H[(long)row * 128 + col] = h;
                ps  += h;
                psq += h * h;
            }
        }
        atomicAdd(&lsum[col], ps);
        atomicAdd(&lsumsq[col], psq);
    }
    __syncthreads();
    if (tid < 128) {
        unsafeAtomicAdd(&colsum[tid],   lsum[tid]);
        unsafeAtomicAdd(&colsumsq[tid], lsumsq[tid]);
    }
}

// ---------- fold BN stats into per-column scale/shift ----------
__global__ void stats_kernel(const float* __restrict__ colsum, const float* __restrict__ colsumsq,
                             const float* __restrict__ gamma, const float* __restrict__ beta,
                             float* __restrict__ scale, float* __restrict__ shift, float n) {
    int j = threadIdx.x;           // 128 threads
    float mean = colsum[j] / n;
    float var  = colsumsq[j] / n - mean * mean;
    float rs   = rsqrtf(var + 1e-5f);
    float sc   = gamma[j] * rs;
    scale[j] = sc;
    shift[j] = beta[j] - mean * sc;
}

// ---------- apply BN, emit bf16 (next layer) or f32 (final) ----------
__global__ void norm_bf16_kernel(const float* __restrict__ h, const float* __restrict__ scale,
                                 const float* __restrict__ shift,
                                 unsigned short* __restrict__ out, long total) {
    long t = (long)blockIdx.x * blockDim.x + threadIdx.x;
    if (t >= total) return;
    int j = (int)(t & 127);
    out[t] = f2bf(h[t] * scale[j] + shift[j]);
}

__global__ void norm_f32_kernel(const float* __restrict__ h, const float* __restrict__ scale,
                                const float* __restrict__ shift,
                                float* __restrict__ out, long total) {
    long t = (long)blockIdx.x * blockDim.x + threadIdx.x;
    if (t >= total) return;
    int j = (int)(t & 127);
    out[t] = h[t] * scale[j] + shift[j];
}

extern "C" void kernel_launch(void* const* d_in, const int* in_sizes, int n_in,
                              void* d_out, int out_size, void* d_ws, size_t ws_size,
                              hipStream_t stream) {
    const int N_ = 100000, E_ = 640000;

    const float* x          = (const float*)d_in[0];
    const int*   edge_index = (const int*)  d_in[1];   // [2,E]; row 0 = src
    const float* edge_attr  = (const float*)d_in[2];
    const float* u          = (const float*)d_in[3];
    const int*   batch      = (const int*)  d_in[4];
    const float* W0 = (const float*)d_in[5];
    const float* b0 = (const float*)d_in[6];
    const float* W1 = (const float*)d_in[7];
    const float* b1 = (const float*)d_in[8];
    const float* W2 = (const float*)d_in[9];
    const float* b2 = (const float*)d_in[10];
    const float* g0  = (const float*)d_in[11];
    const float* be0 = (const float*)d_in[12];
    const float* g1  = (const float*)d_in[13];
    const float* be1 = (const float*)d_in[14];
    const float* g2  = (const float*)d_in[15];
    const float* be2 = (const float*)d_in[16];

    // workspace layout (H aliases sums; bf16 activations alias comb)
    char* ws = (char*)d_ws;
    float*          sums  = (float*)ws;                                    // N*128 f32
    float*          H     = sums;                                          // alias
    float*          cnt   = (float*)(ws + (size_t)N_ * 128 * 4);           // N f32
    unsigned short* comb  = (unsigned short*)(ws + (size_t)N_ * 128 * 4 + (size_t)N_ * 4); // N*384 bf16
    unsigned short* abf   = comb;                                          // alias, N*128 bf16
    unsigned short* wswz0 = (unsigned short*)((char*)comb + (size_t)N_ * 384 * 2);
    unsigned short* wswz1 = wswz0 + 12 * 4096;   // W0 uses 12 K-chunks
    unsigned short* wswz2 = wswz1 + 4  * 4096;   // W1 uses 4
    float* stats = (float*)((char*)wswz0 + (size_t)(12 + 4 + 4) * 4096 * 2);
    float* colsum0 = stats;        float* colsumsq0 = stats + 128;
    float* colsum1 = stats + 256;  float* colsumsq1 = stats + 384;
    float* colsum2 = stats + 512;  float* colsumsq2 = stats + 640;
    float* scale0  = stats + 768;  float* shift0    = stats + 896;
    float* scale1  = stats + 1024; float* shift1    = stats + 1152;
    float* scale2  = stats + 1280; float* shift2    = stats + 1408;

    // 1) zero sums+cnt (contiguous) and BN stat accumulators
    {
        long n = (long)N_ * 128 + N_;
        zero_kernel<<<(int)((n + 255) / 256), 256, 0, stream>>>(sums, n);
        zero_kernel<<<3, 256, 0, stream>>>(stats, 768);
    }

    // 2) weight swizzle into WMMA B-fragment layout
    wswz_kernel<<<(12 * 4096 + 255) / 256, 256, 0, stream>>>(W0, wswz0, 384);
    wswz_kernel<<<(4  * 4096 + 255) / 256, 256, 0, stream>>>(W1, wswz1, 128);
    wswz_kernel<<<(4  * 4096 + 255) / 256, 256, 0, stream>>>(W2, wswz2, 128);

    // 3) scatter-add edge features + counts
    scatter_kernel<<<(E_ * 4 + 255) / 256, 256, 0, stream>>>(edge_index, edge_attr, sums, cnt, E_);

    // 4) build comb (bf16)
    {
        long n = (long)N_ * 384;
        comb_kernel<<<(int)((n + 255) / 256), 256, 0, stream>>>(x, sums, cnt, u, batch, comb, N_);
    }

    int gemm_blocks = (N_ + 127) / 128;
    long nelt = (long)N_ * 128;
    int  nblk = (int)((nelt + 255) / 256);

    // layer 0: 384 -> 128
    gemm_bn_kernel<<<gemm_blocks, 256, 0, stream>>>(comb, 384, wswz0, b0, H, colsum0, colsumsq0, N_);
    stats_kernel<<<1, 128, 0, stream>>>(colsum0, colsumsq0, g0, be0, scale0, shift0, (float)N_);
    norm_bf16_kernel<<<nblk, 256, 0, stream>>>(H, scale0, shift0, abf, nelt);

    // layer 1: 128 -> 128
    gemm_bn_kernel<<<gemm_blocks, 256, 0, stream>>>(abf, 128, wswz1, b1, H, colsum1, colsumsq1, N_);
    stats_kernel<<<1, 128, 0, stream>>>(colsum1, colsumsq1, g1, be1, scale1, shift1, (float)N_);
    norm_bf16_kernel<<<nblk, 256, 0, stream>>>(H, scale1, shift1, abf, nelt);

    // layer 2: 128 -> 128
    gemm_bn_kernel<<<gemm_blocks, 256, 0, stream>>>(abf, 128, wswz2, b2, H, colsum2, colsumsq2, N_);
    stats_kernel<<<1, 128, 0, stream>>>(colsum2, colsumsq2, g2, be2, scale2, shift2, (float)N_);
    norm_f32_kernel<<<nblk, 256, 0, stream>>>(H, scale2, shift2, (float*)d_out, nelt);
}
